// WaveletEmbedding_60705067761821
// MI455X (gfx1250) — compile-verified, matrix operand order checked by reading, not run on hardware
//
#include <hip/hip_runtime.h>

// out(B*S=131072, 256) = x(131072,256) @ W'^T(256,256) + b
// W' = row-wise inverse Haar (waverec) of W, since the Haar operator is orthogonal.
// GEMM in bf16 hi/lo 3-product split with f32 WMMA accumulation (~fp32 accuracy).
// W' fragment tables (256 KB) staged into LDS via the Tensor Data Mover (TDM).

typedef __bf16        v16bf __attribute__((ext_vector_type(16)));
typedef float         v8f   __attribute__((ext_vector_type(8)));
typedef unsigned int  v8u   __attribute__((ext_vector_type(8)));
typedef unsigned int  v4u32 __attribute__((ext_vector_type(4)));
typedef unsigned int  v8u32 __attribute__((ext_vector_type(8)));

#define INV_SQRT2F 0.70710678118654752440f

__device__ __forceinline__ unsigned short f2bf(float f) {
    union { float f; unsigned u; } v; v.f = f;
    unsigned r = v.u + 0x7FFFu + ((v.u >> 16) & 1u);   // round-to-nearest-even
    return (unsigned short)(r >> 16);
}
__device__ __forceinline__ float bf2f(unsigned short h) {
    union { unsigned u; float f; } v; v.u = ((unsigned)h) << 16;
    return v.f;
}

// ---------------------------------------------------------------------------
// Kernel 1: W'[e][k] = inverse 8-level Haar of row e of W, split into bf16
// hi/lo, scattered into the exact 16x16x32 bf16 B-fragment layout so the GEMM
// kernel can load fragments with plain 128-bit LDS reads.
// B fragment element (k_local in [0,32), n_local in [0,16)):
//   lane     = n_local + 16*((k_local%16) >= 8)
//   vgpr j   = 4*(k_local/16) + ((k_local%8)/2)
//   halfword = k_local & 1
// wfrag layout (u16): hi[ ((tile*8+s)*32+lane)*16 + 2*j + hw ], lo at +65536.
// ---------------------------------------------------------------------------
__global__ void __launch_bounds__(256) build_wfrag(const float* __restrict__ W,
                                                   unsigned short* __restrict__ wfrag)
{
    __shared__ float cbuf[256];
    __shared__ float buf0[256];
    __shared__ float buf1[256];
    const int e = blockIdx.x;
    const int k = threadIdx.x;

    cbuf[k] = W[e * 256 + k];
    __syncthreads();
    if (k == 0) buf0[0] = cbuf[0];            // cA_8
    __syncthreads();

    float* prev = buf0;
    float* cur  = buf1;
#pragma unroll 1
    for (int lev = 0; lev < 8; ++lev) {       // cD_8 at [1], cD_7 at [2,3], ... cD_1 at [128,255]
        const int n = 1 << lev;
        if (k < n) {
            const float a = prev[k];
            const float d = cbuf[n + k];
            cur[2 * k]     = (a + d) * INV_SQRT2F;
            cur[2 * k + 1] = (a - d) * INV_SQRT2F;
        }
        __syncthreads();
        float* tmp = prev; prev = cur; cur = tmp;
    }

    const float val = prev[k];                // W'[e][k]
    const unsigned short hi = f2bf(val);
    const unsigned short lo = f2bf(val - bf2f(hi));

    const int t  = e >> 4, nl = e & 15;       // N tile / column-in-tile (n = e)
    const int s  = k >> 5, kl = k & 31;       // K step / k-in-fragment
    const int k16      = kl & 15;
    const int lanehalf = k16 >> 3;
    const int j        = ((kl >> 4) << 2) | ((k16 & 7) >> 1);
    const int hw       = kl & 1;
    const int lane     = nl | (lanehalf << 4);
    const unsigned idx = ((((unsigned)(t * 8 + s) * 32 + lane) << 3) + j) * 2 + hw;

    wfrag[idx]           = hi;
    wfrag[65536u + idx]  = lo;
}

// ---------------------------------------------------------------------------
// Kernel 2: streaming GEMM. 256 threads = 8 waves; each wave owns a 16-row
// stripe of x and produces all 256 output columns. W' fragments (256 KB,
// hi+lo) are staged into LDS once per workgroup by a single TDM instruction.
// ---------------------------------------------------------------------------
__global__ void __launch_bounds__(256) wavelet_gemm(const float* __restrict__ x,
                                                    const float* __restrict__ bias,
                                                    const unsigned int* __restrict__ wfrag,
                                                    float* __restrict__ out)
{
    extern __shared__ unsigned int lds[];     // 65536 dwords: [0,32768) hi, [32768,65536) lo

    // ---- TDM: one tensor_load_to_lds moves the whole 256 KB table L2 -> LDS.
    // 2D tile of 4-byte elements: 16384 x 4 (= 65536 dwords), contiguous on
    // both sides (tensor_dim0_stride == tile_dim0 == 16384).
    if (threadIdx.x < 32) {                   // wave 0 only (TDM ignores EXEC)
        const unsigned           lds_base = (unsigned)(__SIZE_TYPE__)(void*)lds;
        const unsigned long long ga       = (unsigned long long)(__SIZE_TYPE__)wfrag;

        v4u32 g0;                             // D# group 0 (128b)
        g0[0] = 1u;                                        // count=1, user descriptor
        g0[1] = lds_base;                                  // lds_addr[31:0]
        g0[2] = (unsigned)ga;                              // global_addr[31:0]
        g0[3] = ((unsigned)(ga >> 32) & 0x01FFFFFFu)       // global_addr[56:32]
              | (2u << 30);                                // type = 2 ("image")

        v8u32 g1;                             // D# group 1 (256b)
        g1[0] = 2u << 16;                                  // data_size = 2 (4 bytes)
        g1[1] = 16384u << 16;                              // tensor_dim0[15:0] = 16384
        g1[2] = 4u << 16;                                  // tensor_dim0 hi=0 | tensor_dim1 = 4
        g1[3] = 16384u << 16;                              // tensor_dim1 hi=0 | tile_dim0 = 16384
        g1[4] = 4u;                                        // tile_dim1 = 4, tile_dim2 = 0
        g1[5] = 16384u;                                    // tensor_dim0_stride[31:0] = 16384
        g1[6] = 0u;                                        // stride0 hi | stride1 lo
        g1[7] = 0u;                                        // stride1 hi

        asm volatile("tensor_load_to_lds %0, %1" :: "s"(g0), "s"(g1) : "memory");
        __builtin_amdgcn_s_wait_tensorcnt(0);
    }
    __syncthreads();

    const int lane = threadIdx.x & 31;
    const int wave = threadIdx.x >> 5;
    const int half = lane >> 4;
    const int l15  = lane & 15;
    const long rowBase = (long)blockIdx.x * 128 + wave * 16;
    const float* xrow = x + (rowBase + l15) * 256;

    // Build persistent A fragments (16x32 bf16, hi and lo splits) for all 8 K-steps.
    // Lane layout: lanes 0-15 row M=lane, K pairs {k0..k0+7} in v0..v3 and
    // {k0+16..k0+23} in v4..v7 with k0 = s*32 + half*8.
    v16bf a_hi[8], a_lo[8];
#pragma unroll
    for (int s = 0; s < 8; ++s) {
        const int k0 = s * 32 + half * 8;
        const float4 f0 = *(const float4*)(xrow + k0);
        const float4 f1 = *(const float4*)(xrow + k0 + 4);
        const float4 f2 = *(const float4*)(xrow + k0 + 16);
        const float4 f3 = *(const float4*)(xrow + k0 + 20);
        const float g[16] = { f0.x, f0.y, f0.z, f0.w, f1.x, f1.y, f1.z, f1.w,
                              f2.x, f2.y, f2.z, f2.w, f3.x, f3.y, f3.z, f3.w };
        v8u uh, ul;
#pragma unroll
        for (int j = 0; j < 8; ++j) {
            const unsigned short h0 = f2bf(g[2 * j]);
            const unsigned short h1 = f2bf(g[2 * j + 1]);
            const unsigned short q0 = f2bf(g[2 * j]     - bf2f(h0));
            const unsigned short q1 = f2bf(g[2 * j + 1] - bf2f(h1));
            uh[j] = (unsigned)h0 | ((unsigned)h1 << 16);
            ul[j] = (unsigned)q0 | ((unsigned)q1 << 16);
        }
        a_hi[s] = __builtin_bit_cast(v16bf, uh);
        a_lo[s] = __builtin_bit_cast(v16bf, ul);
    }

    // Fragment fetch helper: step u in [0,128) -> hi at lds[u*256 + lane*8],
    // lo at +32768 dwords.
    const unsigned laneOff = (unsigned)lane << 3;

    // Software pipeline: bias value rotates one N-tile ahead; B fragments
    // rotate one K-step ahead, so ds_load latency hides under the WMMAs.
    float bvcur = bias[l15];                  // bias for tile 0
    v16bf bh, bl;
    {
        const unsigned int* p0 = lds + laneOff;
        bh = *(const v16bf*)p0;
        bl = *(const v16bf*)(p0 + 32768);
    }

#pragma unroll 1
    for (int t = 0; t < 16; ++t) {
        // Prefetch next tile's bias now; its s_wait lands after 24 WMMAs.
        const float bvnext = bias[(((t + 1) & 15) << 4) + l15];

        // Bias folded into accumulator init: every C element of this lane is
        // column n = t*16 + (lane&15).
        v8f c;
#pragma unroll
        for (int i = 0; i < 8; ++i) c[i] = bvcur;

#pragma unroll
        for (int s = 0; s < 8; ++s) {
            const int u = t * 8 + s;
            // Prefetch fragments for step u+1 (clamped: last step reloads 127).
            const int un = (u < 127) ? (u + 1) : 127;
            const unsigned int* pn = lds + ((unsigned)un << 8) + laneOff;
            const v16bf nh = *(const v16bf*)pn;
            const v16bf nl = *(const v16bf*)(pn + 32768);

            // 3-product bf16 split: x_hi*W_hi + x_lo*W_hi + x_hi*W_lo
            c = __builtin_amdgcn_wmma_f32_16x16x32_bf16(false, a_hi[s], false, bh,
                                                        (short)0, c, false, false);
            c = __builtin_amdgcn_wmma_f32_16x16x32_bf16(false, a_lo[s], false, bh,
                                                        (short)0, c, false, false);
            c = __builtin_amdgcn_wmma_f32_16x16x32_bf16(false, a_hi[s], false, bl,
                                                        (short)0, c, false, false);
            bh = nh;
            bl = nl;
        }

        // C/D layout: vgpr i -> row m = half*8 + i, column n = lane&15 (fixed per lane)
        float* obase = out + rowBase * 256 + t * 16 + l15;
#pragma unroll
        for (int i = 0; i < 8; ++i) {
            obase[(long)(half * 8 + i) * 256] = c[i];
        }
        bvcur = bvnext;
    }
}

extern "C" void kernel_launch(void* const* d_in, const int* in_sizes, int n_in,
                              void* d_out, int out_size, void* d_ws, size_t ws_size,
                              hipStream_t stream) {
    const float* x = (const float*)d_in[0];   // (16, 8192, 256) f32
    const float* W = (const float*)d_in[1];   // (256, 256) f32
    const float* b = (const float*)d_in[2];   // (256,) f32
    float* out = (float*)d_out;               // (16, 8192, 256) f32

    unsigned short* wfrag = (unsigned short*)d_ws;   // 256 KB: bf16 hi + lo fragment tables

    build_wfrag<<<256, 256, 0, stream>>>(W, wfrag);

    const size_t ldsBytes = 65536u * 4u;      // 256 KB dynamic LDS (<= 320 KB/WGP)
    (void)hipFuncSetAttribute(reinterpret_cast<const void*>(&wavelet_gemm),
                              hipFuncAttributeMaxDynamicSharedMemorySize,
                              (int)ldsBytes);

    // 131072 rows / (8 waves * 16 rows) = 1024 workgroups
    wavelet_gemm<<<1024, 256, ldsBytes, stream>>>(x, b, (const unsigned int*)d_ws, out);
}
